// CausalSelfAttention_12644383719622
// MI455X (gfx1250) — matrix operand order repaired and checked by reading, not compiled, not run on hardware
//
#include <hip/hip_runtime.h>
#include <hip/hip_bf16.h>

#define T_SEQ   2048
#define D_MODEL 1024
#define NHEAD   16
#define HDIM    64

typedef __attribute__((ext_vector_type(16))) __bf16 v16bf;
typedef __attribute__((ext_vector_type(8)))  float  v8f;

union Frag32 { uint4 u[2]; v16bf v; };   // 32 bytes = one 16x32-bf16 A/B fragment per lane

__device__ __forceinline__ unsigned short f2bf(float f) {
  unsigned u = __float_as_uint(f);
  unsigned r = u + 0x7FFFu + ((u >> 16) & 1u);   // round-to-nearest-even
  return (unsigned short)(r >> 16);
}

// async global->LDS copy of 16 bytes per lane (CDNA5 ASYNCcnt path)
__device__ __forceinline__ void async_ld16(const unsigned short* g, unsigned lds_off) {
  asm volatile("global_load_async_to_lds_b128 %0, %1, off"
               :: "v"(lds_off), "v"(g) : "memory");
}
__device__ __forceinline__ void wait_async0() {
  asm volatile("s_wait_asynccnt 0x0" ::: "memory");
}
__device__ __forceinline__ void wait_ds0() {
  asm volatile("s_wait_dscnt 0x0" ::: "memory");
}

// ---------------------------------------------------------------- cast kernel
__global__ __launch_bounds__(256) void cast_f32_to_bf16(
    const float* __restrict__ src, unsigned short* __restrict__ dst, int n4) {
  int i = (blockIdx.x * blockDim.x + threadIdx.x);
  if (i < n4) {
    float4 f = ((const float4*)src)[i];
    ushort4 o;
    o.x = f2bf(f.x); o.y = f2bf(f.y); o.z = f2bf(f.z); o.w = f2bf(f.w);
    ((ushort4*)dst)[i] = o;
  }
}

// ------------------------------------------------- generic bf16 GEMM (A * B^T)
// C[M,N] = A[M,K] * B[N,K]^T, A/B bf16 row-major, C f32. Compile-time dims so
// the writeback collapses to base + 24-bit immediate offsets.
// Block tile 128x64, 256 threads = 8 waves in 4x2 grid, wave tile 32x32.
// A tile staged via GLOBAL_LOAD_ASYNC_TO_LDS_B128; B tile transposed via LDS.
template<int M, int N, int K>
__global__ __launch_bounds__(256) void gemm_bf16_nt(
    const unsigned short* __restrict__ A,
    const unsigned short* __restrict__ B,
    float* __restrict__ C) {
  __shared__ __align__(16) unsigned short As[128][40];  // [m][k], pitch 80B
  __shared__ __align__(16) unsigned short Bt[32][72];   // [k][n] transposed, pitch 144B

  const int tid  = threadIdx.x;
  const int wave = tid >> 5;
  const int lane = tid & 31;
  const int m0   = blockIdx.y * 128;
  const int n0   = blockIdx.x * 64;
  const int wm   = wave >> 1;   // 0..3
  const int wn   = wave & 1;    // 0..1

  const v8f vzero = {0.f,0.f,0.f,0.f,0.f,0.f,0.f,0.f};
  v8f acc[2][2];
  for (int i = 0; i < 2; i++) for (int j = 0; j < 2; j++) acc[i][j] = vzero;

  const int ar  = tid >> 1;          // 0..127
  const int ac  = (tid & 1) * 16;    // 0 or 16
  const int bn  = tid & 63;          // 0..63
  const int bk0 = (tid >> 6) * 8;    // 0,8,16,24
  const unsigned as0 = (unsigned)(size_t)&As[ar][ac];
  const unsigned as1 = (unsigned)(size_t)&As[ar][ac + 8];
  const unsigned short* agBase = A + (size_t)(m0 + ar) * K + ac;
  const unsigned short* bgBase = B + (size_t)(n0 + bn) * K + bk0;

  for (int kb = 0; kb < K; kb += 32) {
    // stage A tile via async DMA (no VGPR round-trip)
    async_ld16(agBase + kb,     as0);
    async_ld16(agBase + kb + 8, as1);
    // warm next B slice in cache while we work on this one
    if (kb + 32 < K) __builtin_prefetch(bgBase + kb + 32, 0, 3);
    // stage B tile transposed: Bt[k][n] = B[n0+n, kb+k]
    uint4 bw = *(const uint4*)(bgBase + kb);
    const unsigned short* bws = (const unsigned short*)&bw;
    #pragma unroll
    for (int i = 0; i < 8; i++) Bt[bk0 + i][bn] = bws[i];
    wait_async0();
    __syncthreads();

    const int ml   = lane & 15;
    const int koff = (lane < 16) ? 0 : 8;   // A-frag half-lane K split
    Frag32 a[2], b[2];
    #pragma unroll
    for (int i = 0; i < 2; i++) {
      const int row = wm * 32 + i * 16 + ml;
      a[i].u[0] = *(const uint4*)&As[row][koff];
      a[i].u[1] = *(const uint4*)&As[row][koff + 16];
      const int col = wn * 32 + i * 16;
      #pragma unroll
      for (int s = 0; s < 8; s++) {
        ((unsigned short*)&b[i])[s]     = Bt[koff + s][col + ml];
        ((unsigned short*)&b[i])[s + 8] = Bt[koff + 16 + s][col + ml];
      }
    }
    #pragma unroll
    for (int i = 0; i < 2; i++)
      #pragma unroll
      for (int j = 0; j < 2; j++)
        acc[i][j] = __builtin_amdgcn_wmma_f32_16x16x32_bf16(
            false, a[i].v, false, b[j].v, (short)0, acc[i][j], false, false);
    __syncthreads();
  }

  // writeback: C layout lane=N col, VGPR=M row (+8 for upper half-lanes).
  // Compile-time N => one base pointer + immediate offsets for all 32 stores.
  const int ml = lane & 15;
  const int rb = (lane < 16) ? 0 : 8;
  float* cp = C + (size_t)(m0 + wm * 32 + rb) * N + n0 + wn * 32 + ml;
  #pragma unroll
  for (int i = 0; i < 2; i++)
    #pragma unroll
    for (int j = 0; j < 2; j++)
      #pragma unroll
      for (int r = 0; r < 8; r++)
        cp[(i * 16 + r) * N + j * 16] = acc[i][j][r];
}

// ----------------------------------- epilogue: v-mix, rms-norm, rotary -> bf16
__global__ __launch_bounds__(256) void qkv_epilogue(
    const float* __restrict__ qf, const float* __restrict__ kf,
    const float* __restrict__ vf, const float* __restrict__ vi,
    const float* __restrict__ lambdas,
    unsigned short* __restrict__ qb, unsigned short* __restrict__ kb,
    unsigned short* __restrict__ vb) {
  const int wave = threadIdx.x >> 5;
  const int lane = threadIdx.x & 31;
  const int idx  = blockIdx.x * 8 + wave;  // t*H + h
  const int t    = idx / NHEAD;
  const int h    = idx % NHEAD;
  const size_t base = (size_t)t * D_MODEL + h * HDIM;

  const float l0 = lambdas[0], l1 = lambdas[1];
  vb[base + lane]      = f2bf(l0 * vf[base + lane]      + l1 * vi[base + lane]);
  vb[base + lane + 32] = f2bf(l0 * vf[base + lane + 32] + l1 * vi[base + lane + 32]);

  // rotary angle: lane j<16 carries theta = t * (1/1024)^(j/15); lanes>=16 passthrough
  float c, s;
  if (lane < 16) {
    float inv = __powf(1.0f / 1024.0f, (float)lane * (1.0f / 15.0f));
    float th  = (float)t * inv;
    c = __cosf(th); s = __sinf(th);
  } else { c = 1.0f; s = 0.0f; }

  #pragma unroll
  for (int which = 0; which < 2; which++) {
    const float* src = which ? kf : qf;
    unsigned short* dst = which ? kb : qb;
    float x1 = src[base + lane], x2 = src[base + lane + 32];
    float ss = x1 * x1 + x2 * x2;
    #pragma unroll
    for (int o = 16; o > 0; o >>= 1) ss += __shfl_xor(ss, o, 32);
    float r = rsqrtf(ss * (1.0f / 64.0f) + 1.1920929e-7f);
    x1 *= r; x2 *= r;
    dst[base + lane]      = f2bf( x1 * c + x2 * s);
    dst[base + lane + 32] = f2bf(-x1 * s + x2 * c);
  }
}

// --------------------------------------------------------- flash attention
// Transposed-score scheme: S^T(s,m) = K x q^T. K and V fragments come straight
// from global memory (row-major == fragment-friendly); q is transposed once per
// wave into private LDS. No __syncthreads in the key loop — waves independent.
// 4 waves/block, each wave owns 16 query rows; block = 64 rows of one head.
__global__ __launch_bounds__(128) void flash_attn(
    const unsigned short* __restrict__ qb, const unsigned short* __restrict__ kb,
    const unsigned short* __restrict__ vb, unsigned short* __restrict__ yb) {
  __shared__ __align__(16) unsigned short qT[4][64][16];  // [wave][d][m]
  __shared__ __align__(16) unsigned short pS[4][16][40];  // [wave][m][s]

  const int tid  = threadIdx.x;
  const int wave = tid >> 5;
  const int lane = tid & 31;
  const int ml   = lane & 15;
  const int lh   = lane >> 4;       // half-lane id
  const int rl0  = lh * 8;          // C-layout row offset for this half
  const int koff = lh * 8;          // A-frag K offset for this half
  const int h    = blockIdx.y;
  const int qt   = blockIdx.x;
  const int qr0  = qt * 64 + wave * 16;
  const size_t hb = (size_t)h * HDIM;

  // ---- stage q transposed (once): qT[d][m] for this wave's 16 rows
  {
    const unsigned short* qp = qb + (size_t)(qr0 + ml) * D_MODEL + hb + lh * 32;
    uint4 cc[4];
    #pragma unroll
    for (int i = 0; i < 4; i++) cc[i] = ((const uint4*)qp)[i];
    const unsigned short* es = (const unsigned short*)cc;
    #pragma unroll
    for (int i = 0; i < 32; i++) qT[wave][lh * 32 + i][ml] = es[i];
  }
  wait_ds0();

  // q as B-fragments (lane = m column, d in table order), kept for the whole loop
  Frag32 qbf[2];
  #pragma unroll
  for (int kc = 0; kc < 2; kc++) {
    #pragma unroll
    for (int s = 0; s < 8; s++) {
      ((unsigned short*)&qbf[kc])[s]     = qT[wave][kc * 32 + koff + s][ml];
      ((unsigned short*)&qbf[kc])[s + 8] = qT[wave][kc * 32 + koff + 16 + s][ml];
    }
  }

  const v8f vzero = {0.f,0.f,0.f,0.f,0.f,0.f,0.f,0.f};
  v8f yacc[4];
  #pragma unroll
  for (int f = 0; f < 4; f++) yacc[f] = vzero;
  float mrow = -3.0e38f, lrow = 0.0f;   // stats for column m = qr0+ml

  const int mg  = qr0 + ml;
  const int nkb = qt * 2 + 2;           // causal: key blocks <= query tile

  for (int kbk = 0; kbk < nkb; kbk++) {
    const int sb = kbk * 32;

    // warm next key/value block in cache
    if (kbk + 1 < nkb) {
      __builtin_prefetch(kb + (size_t)(sb + 32 + lane) * D_MODEL + hb, 0, 3);
      __builtin_prefetch(vb + (size_t)(sb + 32 + lane) * D_MODEL + hb, 0, 3);
    }

    // K as A-fragments, straight from global (row-major over d)
    Frag32 ka[2][2];   // [s-tile][d-chunk]
    #pragma unroll
    for (int st = 0; st < 2; st++) {
      const unsigned short* kp = kb + (size_t)(sb + st * 16 + ml) * D_MODEL + hb;
      #pragma unroll
      for (int kc = 0; kc < 2; kc++) {
        ka[st][kc].u[0] = *(const uint4*)(kp + kc * 32 + koff);
        ka[st][kc].u[1] = *(const uint4*)(kp + kc * 32 + koff + 16);
      }
    }
    // S^T tiles: rows = s, cols = m
    v8f sc[2] = {vzero, vzero};
    #pragma unroll
    for (int st = 0; st < 2; st++)
      #pragma unroll
      for (int kc = 0; kc < 2; kc++)
        sc[st] = __builtin_amdgcn_wmma_f32_16x16x32_bf16(
            false, ka[st][kc].v, false, qbf[kc].v, (short)0, sc[st], false, false);

    // scale + causal mask + online softmax along s (VGPRs + one lane-pair swap)
    float vmax = -3.0e38f;
    #pragma unroll
    for (int st = 0; st < 2; st++)
      #pragma unroll
      for (int r = 0; r < 8; r++) {
        float s = sc[st][r] * 0.125f;
        const int sg = sb + st * 16 + rl0 + r;
        if (sg > mg) s = -3.0e38f;
        sc[st][r] = s;
        vmax = fmaxf(vmax, s);
      }
    vmax = fmaxf(vmax, __shfl_xor(vmax, 16, 32));
    const float mnew = fmaxf(mrow, vmax);
    const float corr = __expf(mrow - mnew);
    float psum = 0.0f;
    #pragma unroll
    for (int st = 0; st < 2; st++)
      #pragma unroll
      for (int r = 0; r < 8; r++) {
        float p = __expf(sc[st][r] - mnew);
        psum += p;
        pS[wave][ml][st * 16 + rl0 + r] = f2bf(p);
      }
    psum += __shfl_xor(psum, 16, 32);
    lrow = lrow * corr + psum;
    mrow = mnew;
    wait_ds0();

    // rescale y accumulators: per-VGPR row m = rl0+j, stats live in lane rl0+j
    #pragma unroll
    for (int j = 0; j < 8; j++) {
      const float cj = __shfl(corr, rl0 + j, 32);
      #pragma unroll
      for (int f = 0; f < 4; f++) yacc[f][j] *= cj;
    }

    // P back as A-fragment (lane = m, s in table order)
    Frag32 pa;
    pa.u[0] = *(const uint4*)&pS[wave][ml][koff];
    pa.u[1] = *(const uint4*)&pS[wave][ml][koff + 16];

    // V as B-fragments straight from global; y += P x V
    #pragma unroll
    for (int f = 0; f < 4; f++) {
      Frag32 vfr;
      const unsigned short* vp = vb + (size_t)sb * D_MODEL + hb + f * 16 + ml;
      #pragma unroll
      for (int s = 0; s < 8; s++) {
        ((unsigned short*)&vfr)[s]     = vp[(size_t)(koff + s) * D_MODEL];
        ((unsigned short*)&vfr)[s + 8] = vp[(size_t)(koff + 16 + s) * D_MODEL];
      }
      yacc[f] = __builtin_amdgcn_wmma_f32_16x16x32_bf16(
          false, pa.v, false, vfr.v, (short)0, yacc[f], false, false);
    }
  }

  // normalize rows and store y (bf16)
  float lj[8];
  #pragma unroll
  for (int j = 0; j < 8; j++) lj[j] = __shfl(lrow, rl0 + j, 32);
  #pragma unroll
  for (int f = 0; f < 4; f++)
    #pragma unroll
    for (int r = 0; r < 8; r++)
      yb[(size_t)(qr0 + rl0 + r) * D_MODEL + hb + f * 16 + ml] =
          f2bf(yacc[f][r] / lj[r]);
}

// ------------------------------------------------------------------ launcher
extern "C" void kernel_launch(void* const* d_in, const int* in_sizes, int n_in,
                              void* d_out, int out_size, void* d_ws, size_t ws_size,
                              hipStream_t stream) {
  const float* x   = (const float*)d_in[0];
  const float* vi  = (const float*)d_in[1];
  const float* Wq  = (const float*)d_in[2];
  const float* Wk  = (const float*)d_in[3];
  const float* Wv  = (const float*)d_in[4];
  const float* Wp  = (const float*)d_in[5];
  const float* lam = (const float*)d_in[6];
  float* out = (float*)d_out;

  const size_t TD = (size_t)T_SEQ * D_MODEL;
  const size_t DD = (size_t)D_MODEL * D_MODEL;

  char* p = (char*)d_ws;
  unsigned short* xb  = (unsigned short*)p; p += TD * 2;
  unsigned short* wqb = (unsigned short*)p; p += DD * 2;
  unsigned short* wkb = (unsigned short*)p; p += DD * 2;
  unsigned short* wvb = (unsigned short*)p; p += DD * 2;
  unsigned short* wpb = (unsigned short*)p; p += DD * 2;
  float* qf = (float*)p; p += TD * 4;
  float* kf = (float*)p; p += TD * 4;
  float* vf = (float*)p; p += TD * 4;
  unsigned short* qbb = (unsigned short*)p; p += TD * 2;
  unsigned short* kbb = (unsigned short*)p; p += TD * 2;
  unsigned short* vbb = (unsigned short*)p; p += TD * 2;
  unsigned short* ybb = (unsigned short*)p; p += TD * 2;

  // 1) cast activations/weights to bf16
  cast_f32_to_bf16<<<(int)(TD / 4 / 256), 256, 0, stream>>>(x,  xb,  (int)(TD / 4));
  cast_f32_to_bf16<<<(int)(DD / 4 / 256), 256, 0, stream>>>(Wq, wqb, (int)(DD / 4));
  cast_f32_to_bf16<<<(int)(DD / 4 / 256), 256, 0, stream>>>(Wk, wkb, (int)(DD / 4));
  cast_f32_to_bf16<<<(int)(DD / 4 / 256), 256, 0, stream>>>(Wv, wvb, (int)(DD / 4));
  cast_f32_to_bf16<<<(int)(DD / 4 / 256), 256, 0, stream>>>(Wp, wpb, (int)(DD / 4));

  // 2) QKV projections: [2048,1024] @ [1024,1024]^T
  dim3 gg(D_MODEL / 64, T_SEQ / 128);
  gemm_bf16_nt<T_SEQ, D_MODEL, D_MODEL><<<gg, 256, 0, stream>>>(xb, wqb, qf);
  gemm_bf16_nt<T_SEQ, D_MODEL, D_MODEL><<<gg, 256, 0, stream>>>(xb, wkb, kf);
  gemm_bf16_nt<T_SEQ, D_MODEL, D_MODEL><<<gg, 256, 0, stream>>>(xb, wvb, vf);

  // 3) epilogue (v-mix, rms-norm, rotary) -> bf16 q/k/v
  qkv_epilogue<<<T_SEQ * NHEAD / 8, 256, 0, stream>>>(qf, kf, vf, vi, lam, qbb, kbb, vbb);

  // 4) flash attention
  dim3 fg(T_SEQ / 64, NHEAD);
  flash_attn<<<fg, 128, 0, stream>>>(qbb, kbb, vbb, ybb);

  // 5) output projection -> f32 d_out
  gemm_bf16_nt<T_SEQ, D_MODEL, D_MODEL><<<gg, 256, 0, stream>>>(ybb, wpb, out);
}
